// QuantumLLM_63093069578346
// MI455X (gfx1250) — compile-verified, matrix operand order checked by reading, not run on hardware
//
#include <hip/hip_runtime.h>
#include <math.h>

typedef unsigned short u16;
typedef __attribute__((ext_vector_type(16))) __bf16 v16bf;
typedef __attribute__((ext_vector_type(8)))  float  v8f;
typedef int v4i __attribute__((ext_vector_type(4)));

// ---------------- problem constants ----------------
constexpr int BB = 4;
constexpr int LL = 4096;
constexpr int DD = 512;
constexpr int MM = BB * LL;          // 16384 tokens
constexpr float TWO_PI_F = 6.283185307179586f;

// ---------------- CDNA5 async global->LDS copy (ASYNCcnt path) ----------------
#if defined(__has_builtin)
#  if __has_builtin(__builtin_amdgcn_global_load_async_to_lds_b128) && \
      __has_builtin(__builtin_amdgcn_s_wait_asynccnt)
#    define HAVE_ASYNC_LDS 1
#  endif
#endif
#ifndef HAVE_ASYNC_LDS
#  define HAVE_ASYNC_LDS 0
#endif

#if HAVE_ASYNC_LDS
typedef __attribute__((address_space(1))) v4i gv4i;   // global 128-bit element
typedef __attribute__((address_space(3))) v4i lv4i;   // LDS 128-bit element
#endif

__device__ __forceinline__ void cp_async_b128(u16* ldst, const u16* gsrc) {
#if HAVE_ASYNC_LDS
    gv4i* gp = (gv4i*)(unsigned long long)(uintptr_t)gsrc;
    lv4i* lp = (lv4i*)(unsigned int)(uintptr_t)ldst;   // low 32 bits = LDS offset
    __builtin_amdgcn_global_load_async_to_lds_b128(gp, lp, 0, 0);
#else
    *reinterpret_cast<uint4*>(ldst) = *reinterpret_cast<const uint4*>(gsrc);
#endif
}

__device__ __forceinline__ void cp_async_wait() {
#if HAVE_ASYNC_LDS
    __builtin_amdgcn_s_wait_asynccnt(0);
#endif
}

// ---------------- helpers ----------------
__device__ __forceinline__ u16 f2bf(float f) {
    unsigned int u = __float_as_uint(f);
    u += 0x7FFFu + ((u >> 16) & 1u);          // round-to-nearest-even
    return (u16)(u >> 16);
}

struct alignas(16) Half8 { u16 h[8]; };

__device__ __forceinline__ v16bf make_frag(const u16* p0, const u16* p1) {
    union { v16bf v; Half8 q[2]; } u;
    u.q[0] = *reinterpret_cast<const Half8*>(p0);
    u.q[1] = *reinterpret_cast<const Half8*>(p1);
    return u.v;
}

__device__ __forceinline__ v8f v8zero() {
    v8f z = {0.f,0.f,0.f,0.f,0.f,0.f,0.f,0.f};
    return z;
}

__device__ __forceinline__ v8f wmma_bf16(v16bf a, v16bf b, v8f c) {
    return __builtin_amdgcn_wmma_f32_16x16x32_bf16(false, a, false, b,
                                                   (short)0, c, false, false);
}

// ---------------- generic WMMA GEMM mainloop (double-buffered, async) ----------
// C(M x N) = A(M x K, bf16 row-major) * W(N x K, bf16 row-major)^T
// block = 256 threads (8 waves), tile 64(M) x 128(N), BK = 32.
// wave grid: 4 (M) x 2 (N); each wave -> 16 x 64 (4 N-frags).
// ldsA: 2 * 64*32 u16, ldsB: 2 * 128*32 u16 (double buffered).
constexpr int ASZ = 64 * 32;
constexpr int BSZ = 128 * 32;

__device__ __forceinline__ void gemm_tile(const u16* __restrict__ A,
                                          const u16* __restrict__ W,
                                          int K, int m0, int n0,
                                          u16* ldsA, u16* ldsB,
                                          v8f (&acc)[4]) {
    const int tid   = threadIdx.x;
    const int lane  = tid & 31;
    const int wave  = tid >> 5;
    const int waveM = wave >> 1;      // 0..3
    const int waveN = wave & 1;       // 0..1
    const int half  = lane >> 4;
    const int r     = lane & 15;
    const int kof   = half * 8;

    #pragma unroll
    for (int f = 0; f < 4; ++f) acc[f] = v8zero();

    const int rowA = tid >> 2;              // 0..63
    const int colA = (tid & 3) * 8;         // 0,8,16,24
    const int rowB = tid >> 1;              // 0..127
    const int colB = (tid & 1) * 16;        // 0,16

    const u16* Abase = A + (size_t)(m0 + rowA) * K + colA;
    const u16* Wbase = W + (size_t)(n0 + rowB) * K + colB;
    u16* aDst = ldsA + rowA * 32 + colA;
    u16* bDst = ldsB + rowB * 32 + colB;

    auto stage = [&](int k0, int buf) {
        cp_async_b128(aDst + buf * ASZ,     Abase + k0);
        cp_async_b128(bDst + buf * BSZ,     Wbase + k0);
        cp_async_b128(bDst + buf * BSZ + 8, Wbase + k0 + 8);
    };

    // prologue: stage tile 0 into buffer 0
    stage(0, 0);
    cp_async_wait();
    __syncthreads();

    int cur = 0;
    for (int k0 = 0; k0 < K; k0 += 32) {
        if (k0 + 32 < K) {
            stage(k0 + 32, cur ^ 1);                       // overlap with compute
            __builtin_prefetch(Abase + k0 + 64, 0, 1);     // global_prefetch_b8
            __builtin_prefetch(Wbase + k0 + 64, 0, 1);
        }
        const u16* la = ldsA + cur * ASZ;
        const u16* lb = ldsB + cur * BSZ;
        v16bf a = make_frag(&la[(waveM * 16 + r) * 32 + kof],
                            &la[(waveM * 16 + r) * 32 + kof + 16]);
        #pragma unroll
        for (int f = 0; f < 4; ++f) {
            int rb = (waveN * 64 + f * 16 + r) * 32 + kof;
            v16bf b = make_frag(&lb[rb], &lb[rb + 16]);
            acc[f] = wmma_bf16(a, b, acc[f]);
        }
        cp_async_wait();
        __syncthreads();
        cur ^= 1;
    }
}

// ---------------- kernel 0: f32 -> bf16 convert ----------------
__global__ void k_cvt_bf16(const float* __restrict__ src, u16* __restrict__ dst, int n) {
    int i = blockIdx.x * blockDim.x + threadIdx.x;
    if (i < n) dst[i] = f2bf(src[i]);
}

// ---------------- kernel 1: fold qkv weights ----------------
// W2(2048 x 1024): rows 0..511 -> q_r, 512..1023 -> k_r, 1024..1535 -> v_r, 1536..2047 -> v_i
__global__ void k_fold(const float* __restrict__ qw, const float* __restrict__ qb,
                       u16* __restrict__ w2, float* __restrict__ b2) {
    const int jp = blockIdx.x;                            // 0..2047
    const int j  = (jp < 512) ? jp : ((jp < 1024) ? jp + 512 : jp + 1024);
    const float* row = qw + (size_t)j * 3072;
    for (int c = threadIdx.x; c < 1024; c += blockDim.x) {
        float s;
        if (c < 512) s = row[c]        + row[1024 + c]       + row[2048 + c];
        else { int cc = c - 512; s = row[512 + cc] + row[1536 + cc] + row[2560 + cc]; }
        w2[(size_t)jp * 1024 + c] = f2bf(s);
    }
    if (threadIdx.x == 0) b2[jp] = qb[j];
}

// ---------------- kernel 2: GEMM1 + bias + GELU ----------------
__global__ void k_gemm_ft(const u16* __restrict__ xbf, const u16* __restrict__ ftw,
                          const float* __restrict__ ftb, float* __restrict__ g) {
    __shared__ __align__(16) u16 sA[2 * ASZ];
    __shared__ __align__(16) u16 sB[2 * BSZ];
    const int m0 = blockIdx.y * 64, n0 = blockIdx.x * 128;
    v8f acc[4];
    gemm_tile(xbf, ftw, 512, m0, n0, sA, sB, acc);

    const int tid = threadIdx.x, lane = tid & 31, wave = tid >> 5;
    const int waveM = wave >> 1, waveN = wave & 1;
    const int half = lane >> 4, r = lane & 15;
    #pragma unroll
    for (int f = 0; f < 4; ++f) {
        const int n = n0 + waveN * 64 + f * 16 + r;
        const float bias = ftb[n];
        #pragma unroll
        for (int v = 0; v < 8; ++v) {
            const int m = m0 + waveM * 16 + half * 8 + v;
            float t = acc[f][v] + bias;
            float gg = 0.5f * t * (1.0f + tanhf(0.7978845608028654f *
                                    (t + 0.044715f * t * t * t)));
            g[(size_t)m * 1024 + n] = gg;
        }
    }
}

// ---------------- kernel 3: LayerNorm + phase pipeline ----------------
__global__ void k_ln_phase(const float* __restrict__ g,
                           const float* __restrict__ ln_g, const float* __restrict__ ln_b,
                           const float* __restrict__ exc, u16* __restrict__ ri) {
    const int row = blockIdx.x;
    const int tid = threadIdx.x;
    __shared__ float nb[1024];
    __shared__ float red1[8], red2[8];
    __shared__ float s_mu, s_inv;

    const float* grow = g + (size_t)row * 1024;
    float vals[4];
    float s1 = 0.f, s2 = 0.f;
    #pragma unroll
    for (int j = 0; j < 4; ++j) {
        float x = grow[tid + 256 * j];
        vals[j] = x; s1 += x; s2 += x * x;
    }
    #pragma unroll
    for (int off = 16; off; off >>= 1) {
        s1 += __shfl_xor(s1, off, 32);
        s2 += __shfl_xor(s2, off, 32);
    }
    if ((tid & 31) == 0) { red1[tid >> 5] = s1; red2[tid >> 5] = s2; }
    __syncthreads();
    if (tid == 0) {
        float t1 = 0.f, t2 = 0.f;
        for (int w = 0; w < 8; ++w) { t1 += red1[w]; t2 += red2[w]; }
        float mu  = t1 * (1.0f / 1024.0f);
        float var = t2 * (1.0f / 1024.0f) - mu * mu;
        s_mu  = mu;
        s_inv = rsqrtf(var + 1e-5f);
    }
    __syncthreads();
    const float mu = s_mu, inv = s_inv;
    #pragma unroll
    for (int j = 0; j < 4; ++j) {
        int c = tid + 256 * j;
        nb[c] = (vals[j] - mu) * inv * ln_g[c] + ln_b[c];
    }
    __syncthreads();

    const float e = exc[0];
    #pragma unroll
    for (int j = 0; j < 2; ++j) {
        const int d = tid + 256 * j;                    // 0..511
        float rp = nb[d], ip = nb[d + 512];
        float ang = atan2f(ip + 1e-8f, rp + 1e-8f);
        int ii = (int)(ang * (1023.0f / TWO_PI_F));     // trunc toward zero
        int idx = ((ii % 1024) + 1024) % 1024;
        float th = (float)idx * (TWO_PI_F / 1023.0f);
        float real = rp * __cosf(th);
        float imag = ip * __sinf(th);
        float nrm = rsqrtf(real * real + imag * imag + 1e-8f) * 0.04419417382415922f;
        real *= nrm; imag *= nrm;
        float amp = sqrtf(real * real + imag * imag + 1e-8f);
        float coll = (amp < 0.1f) ? 1.0f : 0.0f;
        float el = (float)d * (TWO_PI_F / 511.0f);
        float elm = fmodf(el, TWO_PI_F);
        int lidx = (int)(elm * (1024.0f / TWO_PI_F));
        lidx = min(max(lidx, 0), 1023);
        float lth = (float)lidx * (TWO_PI_F / 1023.0f);
        real += coll * __cosf(lth) * e;
        imag += coll * __sinf(lth) * e;
        float nrm2 = rsqrtf(real * real + imag * imag + 1e-8f);
        real *= nrm2; imag *= nrm2;
        ri[(size_t)row * 1024 + d]       = f2bf(real);
        ri[(size_t)row * 1024 + 512 + d] = f2bf(imag);
    }
}

// ---------------- kernel 4: folded QKV GEMM, scatter q / k / v^T ----------------
__global__ void k_gemm_qkv(const u16* __restrict__ ri, const u16* __restrict__ w2,
                           const float* __restrict__ b2,
                           u16* __restrict__ qb, u16* __restrict__ kb,
                           u16* __restrict__ vT) {
    __shared__ __align__(16) u16 sA[2 * ASZ];
    __shared__ __align__(16) u16 sB[2 * BSZ];
    const int m0 = blockIdx.y * 64, n0 = blockIdx.x * 128;
    v8f acc[4];
    gemm_tile(ri, w2, 1024, m0, n0, sA, sB, acc);

    const int tid = threadIdx.x, lane = tid & 31, wave = tid >> 5;
    const int waveM = wave >> 1, waveN = wave & 1;
    const int half = lane >> 4, r = lane & 15;
    #pragma unroll
    for (int f = 0; f < 4; ++f) {
        const int n = n0 + waveN * 64 + f * 16 + r;
        const float bias = b2[n];
        #pragma unroll
        for (int v = 0; v < 8; ++v) {
            const int m = m0 + waveM * 16 + half * 8 + v;     // token index
            u16 hv = f2bf(acc[f][v] + bias);
            if (n < 512) {
                qb[(size_t)m * 512 + n] = hv;
            } else if (n < 1024) {
                kb[(size_t)m * 512 + (n - 512)] = hv;
            } else {
                int b = m >> 12, l = m & 4095;
                vT[((size_t)(b * 1024 + (n - 1024))) * 4096 + l] = hv;
            }
        }
    }
}

// ---------------- kernel 5: flash attention ----------------
// grid (L/16, B), 256 threads. WG handles 16 query rows; wave w owns
// key-cols [w*16,+16) for QK^T and output cols [w*128,+128) for PV.
__global__ void k_attn(const u16* __restrict__ qb, const u16* __restrict__ kb,
                       const u16* __restrict__ vT, u16* __restrict__ ob) {
    const int bI = blockIdx.y;
    const int m0 = blockIdx.x * 16;
    const int tid = threadIdx.x, lane = tid & 31, wave = tid >> 5;
    const int half = lane >> 4, r = lane & 15, kof = half * 8;

    __shared__ __align__(16) u16 qt[16 * 512];
    __shared__ __align__(16) u16 pb[16 * 128];
    __shared__ float partmax[8][16], partsum[8][16];
    __shared__ float rowM[16], rowL[16], newM[16], corr[16];

    const u16* qrow = qb + ((size_t)(bI * 4096 + m0)) * 512;
    for (int i = tid * 8; i < 16 * 512; i += 256 * 8)
        cp_async_b128(&qt[i], &qrow[i]);                 // async Q stage
    cp_async_wait();
    if (tid < 16) { rowM[tid] = -1e30f; rowL[tid] = 0.f; }
    __syncthreads();

    v8f o[8];
    #pragma unroll
    for (int f = 0; f < 8; ++f) o[f] = v8zero();
    const float scale = 0.04419417382415922f;   // 512^-0.5

    for (int kc = 0; kc < 4096; kc += 128) {
        // ---- S = Q K^T for this wave's 16 keys ----
        v8f s = v8zero();
        const u16* kbase = kb + ((size_t)(bI * 4096 + kc + wave * 16 + r)) * 512 + kof;
        #pragma unroll
        for (int d0 = 0; d0 < 512; d0 += 32) {
            v16bf a  = make_frag(&qt[r * 512 + d0 + kof], &qt[r * 512 + d0 + kof + 16]);
            v16bf bk = make_frag(kbase + d0, kbase + d0 + 16);
            s = wmma_bf16(a, bk, s);
        }
        // ---- per-wave row max over its 16 key cols ----
        #pragma unroll
        for (int v = 0; v < 8; ++v) {
            float x = s[v] * scale;
            s[v] = x;
            #pragma unroll
            for (int off = 8; off; off >>= 1) x = fmaxf(x, __shfl_xor(x, off, 16));
            if (r == 0) partmax[wave][v + half * 8] = x;
        }
        __syncthreads();
        if (tid < 16) {
            float nm = rowM[tid];
            for (int w = 0; w < 8; ++w) nm = fmaxf(nm, partmax[w][tid]);
            corr[tid] = __expf(rowM[tid] - nm);
            newM[tid] = nm;
            rowM[tid] = nm;
        }
        __syncthreads();
        // ---- P = exp(S - M), rescale O, partial sums ----
        #pragma unroll
        for (int v = 0; v < 8; ++v) {
            const int row = v + half * 8;
            float p = __expf(s[v] - newM[row]);
            pb[row * 128 + wave * 16 + r] = f2bf(p);
            float x = p;
            #pragma unroll
            for (int off = 8; off; off >>= 1) x += __shfl_xor(x, off, 16);
            if (r == 0) partsum[wave][row] = x;
            const float c = corr[row];
            #pragma unroll
            for (int f = 0; f < 8; ++f) o[f][v] *= c;
        }
        __syncthreads();
        if (tid < 16) {
            float sa = 0.f;
            for (int w = 0; w < 8; ++w) sa += partsum[w][tid];
            rowL[tid] = rowL[tid] * corr[tid] + sa;
        }
        // ---- O += P V for this wave's 128 output cols ----
        #pragma unroll
        for (int kk = 0; kk < 128; kk += 32) {
            v16bf pa = make_frag(&pb[r * 128 + kk + kof], &pb[r * 128 + kk + kof + 16]);
            #pragma unroll
            for (int f = 0; f < 8; ++f) {
                const u16* vp = vT + ((size_t)(bI * 1024 + wave * 128 + f * 16 + r)) * 4096
                                + kc + kk + kof;
                v16bf vb = make_frag(vp, vp + 16);
                o[f] = wmma_bf16(pa, vb, o[f]);
            }
        }
        __syncthreads();
    }

    #pragma unroll
    for (int v = 0; v < 8; ++v) {
        const int row = v + half * 8;
        const float invl = 1.0f / rowL[row];
        const size_t base = ((size_t)(bI * 4096 + m0 + row)) * 1024;
        #pragma unroll
        for (int f = 0; f < 8; ++f) {
            const int n = wave * 128 + f * 16 + r;
            ob[base + n] = f2bf(o[f][v] * invl);
        }
    }
}

// ---------------- kernel 6: output GEMM + split store ----------------
__global__ void k_gemm_out(const u16* __restrict__ ob, const u16* __restrict__ opw,
                           const float* __restrict__ opb, float* __restrict__ out) {
    __shared__ __align__(16) u16 sA[2 * ASZ];
    __shared__ __align__(16) u16 sB[2 * BSZ];
    const int m0 = blockIdx.y * 64, n0 = blockIdx.x * 128;
    v8f acc[4];
    gemm_tile(ob, opw, 1024, m0, n0, sA, sB, acc);

    const int tid = threadIdx.x, lane = tid & 31, wave = tid >> 5;
    const int waveM = wave >> 1, waveN = wave & 1;
    const int half = lane >> 4, r = lane & 15;
    const size_t HALF = (size_t)BB * LL * DD;   // 8388608
    #pragma unroll
    for (int f = 0; f < 4; ++f) {
        const int n = n0 + waveN * 64 + f * 16 + r;
        const float bias = opb[n];
        #pragma unroll
        for (int v = 0; v < 8; ++v) {
            const int m = m0 + waveM * 16 + half * 8 + v;
            float t = acc[f][v] + bias;
            if (n < 512) out[(size_t)m * 512 + n] = t;
            else         out[HALF + (size_t)m * 512 + (n - 512)] = t;
        }
    }
}

// ---------------- host launcher ----------------
extern "C" void kernel_launch(void* const* d_in, const int* in_sizes, int n_in,
                              void* d_out, int out_size, void* d_ws, size_t ws_size,
                              hipStream_t stream) {
    const float* x    = (const float*)d_in[0];
    const float* ft_w = (const float*)d_in[1];
    const float* ft_b = (const float*)d_in[2];
    const float* ln_g = (const float*)d_in[3];
    const float* ln_b = (const float*)d_in[4];
    const float* exc  = (const float*)d_in[5];
    const float* qkvw = (const float*)d_in[6];
    const float* qkvb = (const float*)d_in[7];
    const float* op_w = (const float*)d_in[8];
    const float* op_b = (const float*)d_in[9];
    float* out = (float*)d_out;

    char* ws = (char*)d_ws;
    u16*   xbf = (u16*)  (ws);                        // 16 MB
    u16*   ftw = (u16*)  (ws + 16777216);             //  1 MB
    float* g   = (float*)(ws + 17825792);             // 64 MB
    u16*   ri  = (u16*)  (ws + 84934656);             // 32 MB
    u16*   w2  = (u16*)  (ws + 118489088);            //  4 MB
    float* b2  = (float*)(ws + 122683392);            //  8 KB
    u16*   qb  = (u16*)  (ws + 122691584);            // 16 MB
    u16*   kb  = (u16*)  (ws + 139468800);            // 16 MB
    u16*   vT  = (u16*)  (ws + 156246016);            // 32 MB
    u16*   ob  = (u16*)  (ws + 189800448);            // 32 MB
    u16*   opw = (u16*)  (ws + 223354880);            //  2 MB

    const int nX  = MM * DD;          // 8388608
    const int nFW = 1024 * 512;
    const int nOW = 1024 * 1024;
    k_cvt_bf16<<<(nX  + 255) / 256, 256, 0, stream>>>(x,    xbf, nX);
    k_cvt_bf16<<<(nFW + 255) / 256, 256, 0, stream>>>(ft_w, ftw, nFW);
    k_cvt_bf16<<<(nOW + 255) / 256, 256, 0, stream>>>(op_w, opw, nOW);
    k_fold<<<2048, 256, 0, stream>>>(qkvw, qkvb, w2, b2);

    k_gemm_ft <<<dim3(1024 / 128, MM / 64), 256, 0, stream>>>(xbf, ftw, ft_b, g);
    k_ln_phase<<<MM, 256, 0, stream>>>(g, ln_g, ln_b, exc, ri);
    k_gemm_qkv<<<dim3(2048 / 128, MM / 64), 256, 0, stream>>>(ri, w2, b2, qb, kb, vT);
    k_attn    <<<dim3(LL / 16, BB), 256, 0, stream>>>(qb, kb, vT, ob);
    k_gemm_out<<<dim3(1024 / 128, MM / 64), 256, 0, stream>>>(ob, opw, op_b, out);
}
